// Decoder_67430986547259
// MI455X (gfx1250) — matrix operand order repaired and checked by reading, not compiled, not run on hardware
//
#include <hip/hip_runtime.h>
#include <cstdint>

// ---------------------------------------------------------------------------
// CDNA5 (gfx1250, wave32) transformer block: RMSNorm -> MQA attn -> Wo+res ->
// RMSNorm -> GELU FFN + res.  All GEMMs / attention matmuls use
// v_wmma_f32_16x16x32_bf16 (f32 accumulate).  Double-buffered LDS staging,
// one barrier per K-step, packed-b64 transposed staging, ones-column rowsum.
// ---------------------------------------------------------------------------

typedef __attribute__((ext_vector_type(16))) __bf16        v16bf;
typedef __attribute__((ext_vector_type(8)))  float         v8f;
typedef __attribute__((ext_vector_type(4)))  unsigned int  v4u;
typedef __attribute__((ext_vector_type(2)))  unsigned int  v2u;

#define D_MODEL 4096
#define N_HEADS 32
#define D_KH    128
#define D_FF    16384
#define SEQ     2048
#define NTOK    4096   // 2 * 2048 tokens

union ABFrag { v16bf v; v4u q[2]; };
union Pack4  { v2u d; unsigned short u[4]; };   // 4 bf16 = one b64

__device__ __forceinline__ unsigned short f2bf(float f) {
    unsigned int u = __float_as_uint(f);
    u += 0x7FFFu + ((u >> 16) & 1u);          // round to nearest even
    return (unsigned short)(u >> 16);
}

__device__ __forceinline__ v8f wmma_bf16(const ABFrag& a, const ABFrag& b, v8f c) {
    return __builtin_amdgcn_wmma_f32_16x16x32_bf16(false, a.v, false, b.v,
                                                   (short)0, c, false, false);
}

// ---------------------------------------------------------------------------
// fp32 -> bf16 bulk conversion (vectorized, n must be a multiple of 4)
// ---------------------------------------------------------------------------
__global__ void cvt_f32_to_bf16(const float* __restrict__ src,
                                unsigned short* __restrict__ dst, int n4) {
    int i = blockIdx.x * blockDim.x + threadIdx.x;
    if (i < n4) {
        float4 f = reinterpret_cast<const float4*>(src)[i];
        unsigned long long p =
            (unsigned long long)f2bf(f.x)        |
            ((unsigned long long)f2bf(f.y) << 16) |
            ((unsigned long long)f2bf(f.z) << 32) |
            ((unsigned long long)f2bf(f.w) << 48);
        reinterpret_cast<unsigned long long*>(dst)[i] = p;
    }
}

// ---------------------------------------------------------------------------
// RMSNorm: one block per token row, output scale * x / rms as bf16
// ---------------------------------------------------------------------------
__global__ void rmsnorm_to_bf16(const float* __restrict__ x,
                                const float* __restrict__ scale,
                                unsigned short* __restrict__ out) {
    __shared__ float red[256];
    const int row = blockIdx.x;
    const float* xr = x + (size_t)row * D_MODEL;
    float ss = 0.f;
    for (int c = threadIdx.x; c < D_MODEL; c += 256) { float v = xr[c]; ss += v * v; }
    red[threadIdx.x] = ss;
    __syncthreads();
    for (int s = 128; s > 0; s >>= 1) {
        if ((int)threadIdx.x < s) red[threadIdx.x] += red[threadIdx.x + s];
        __syncthreads();
    }
    const float rinv = 1.f / sqrtf(red[0] * (1.f / D_MODEL) + 1e-10f);
    unsigned short* orow = out + (size_t)row * D_MODEL;
    for (int c = threadIdx.x; c < D_MODEL; c += 256)
        orow[c] = f2bf(xr[c] * rinv * scale[c]);
}

// ---------------------------------------------------------------------------
// bf16 WMMA GEMM: C[M,N] = A[M,K] @ B[K,N] + bias  (A,B bf16 row-major)
// OUT_F32_RES=1 : outF[idx] = res[idx] + acc + bias   (fp32, residual fused)
// OUT_F32_RES=0 : outB = bf16( [gelu](acc + bias) )
// 256 threads (8 waves, 2x4), block tile 128x128, wave tile 64x32, K-step 32.
// Double-buffered LDS (A row-major pad 40; B transposed Bt[n][k] pad 40) so
// operand fragments are 16B-aligned, bank-conflict-free ds_load_b128 pairs.
// B transpose staged as 4x4 micro-tiles -> packed ds_store_b64 (k-pairs are
// contiguous in Bt).  Software pipeline: regs prefetch next tile during WMMA;
// global_prefetch for tile s+2.
// ---------------------------------------------------------------------------
template<int OUT_F32_RES, int DO_GELU>
__global__ __launch_bounds__(256, 1)
void gemm_bf16_wmma(const unsigned short* __restrict__ A,
                    const unsigned short* __restrict__ B,
                    const float* __restrict__ bias,
                    const float* __restrict__ res,
                    float* __restrict__ outF,
                    unsigned short* __restrict__ outB,
                    int M, int N, int K) {
    __shared__ unsigned short As[2][128 * 40];
    __shared__ unsigned short Bs[2][128 * 40];

    const int tid  = threadIdx.x;
    const int lane = tid & 31, wave = tid >> 5;
    const int half = lane >> 4, l15 = lane & 15;
    const int wm = wave & 1, wn = wave >> 1;              // 2 (M) x 4 (N) waves
    const int bm = blockIdx.y * 128, bn = blockIdx.x * 128;

    v8f acc[4][2];
    for (int i = 0; i < 4; i++)
        for (int j = 0; j < 2; j++)
            for (int r = 0; r < 8; r++) acc[i][j][r] = 0.f;

    const int arow = tid >> 1, aseg = tid & 1;  // A staging: 2 thr/row, 16 elem
    const int bnq  = tid & 31, bkq = tid >> 5;  // B staging: 4(n) x 4(k) / thr

    v4u  a0, a1;          // in-flight A (32B / thread)
    Pack4 brow[4];        // in-flight B 4x4 micro-tile (32B / thread)

    const unsigned short* Abase = A + (size_t)(bm + arow) * K + aseg * 16;
    const unsigned short* Bbase = B + bn + bnq * 4;       // + (k)*N per row

    auto loadTile = [&](int kt) {
        const v4u* ga = reinterpret_cast<const v4u*>(Abase + kt);
        a0 = ga[0]; a1 = ga[1];
        #pragma unroll
        for (int kk = 0; kk < 4; kk++)
            brow[kk].d = *reinterpret_cast<const v2u*>(
                Bbase + (size_t)(kt + bkq * 4 + kk) * N);
    };
    auto storeTile = [&](int buf) {
        v4u* sA = reinterpret_cast<v4u*>(&As[buf][arow * 40 + aseg * 16]);
        sA[0] = a0; sA[1] = a1;
        #pragma unroll
        for (int i = 0; i < 4; i++) {   // pack 4 k-values (contiguous in Bt row)
            Pack4 w;
            w.u[0] = brow[0].u[i]; w.u[1] = brow[1].u[i];
            w.u[2] = brow[2].u[i]; w.u[3] = brow[3].u[i];
            *reinterpret_cast<v2u*>(&Bs[buf][(bnq * 4 + i) * 40 + bkq * 4]) = w.d;
        }
    };

    loadTile(0);
    storeTile(0);
    __syncthreads();

    const int nsteps = K >> 5;
    for (int s = 0; s < nsteps; s++) {
        const int cur = s & 1;
        const bool hasNext = (s + 1) < nsteps;
        if (hasNext) loadTile((s + 1) << 5);
        if (s + 2 < nsteps) {                 // L2/L0 prefetch two tiles ahead
            const int kp = (s + 2) << 5;
            __builtin_prefetch(Abase + kp, 0, 0);
            __builtin_prefetch(Bbase + (size_t)(kp + bkq * 4) * N, 0, 0);
        }

        // operand fragments from current LDS buffer
        ABFrag af[4], bf[2];
        #pragma unroll
        for (int mi = 0; mi < 4; mi++) {
            const int m = wm * 64 + mi * 16 + l15;
            af[mi].q[0] = *reinterpret_cast<const v4u*>(&As[cur][m * 40 + 8 * half]);
            af[mi].q[1] = *reinterpret_cast<const v4u*>(&As[cur][m * 40 + 16 + 8 * half]);
        }
        #pragma unroll
        for (int ni = 0; ni < 2; ni++) {
            const int n = wn * 32 + ni * 16 + l15;
            const v4u* p = reinterpret_cast<const v4u*>(&Bs[cur][n * 40 + 8 * half]);
            bf[ni].q[0] = p[0]; bf[ni].q[1] = p[2];
        }
        #pragma unroll
        for (int mi = 0; mi < 4; mi++)
            #pragma unroll
            for (int ni = 0; ni < 2; ni++)
                acc[mi][ni] = wmma_bf16(af[mi], bf[ni], acc[mi][ni]);

        if (hasNext) storeTile(cur ^ 1);
        __syncthreads();
    }

    // epilogue
    #pragma unroll
    for (int mi = 0; mi < 4; mi++)
        #pragma unroll
        for (int ni = 0; ni < 2; ni++) {
            const int col = bn + wn * 32 + ni * 16 + l15;
            const float bval = bias[col];
            #pragma unroll
            for (int r = 0; r < 8; r++) {
                const int row = bm + wm * 64 + mi * 16 + r + 8 * half;
                float v = acc[mi][ni][r] + bval;
                const size_t idx = (size_t)row * N + col;
                if (OUT_F32_RES) {
                    outF[idx] = res[idx] + v;
                } else {
                    if (DO_GELU) v = 0.5f * v * (1.f + erff(v * 0.70710678118654752f));
                    outB[idx] = f2bf(v);
                }
            }
        }
}

// ---------------------------------------------------------------------------
// MQA flash attention.  Block = (batch b, head h, 128-query tile); 8 waves,
// each owning 16 queries.  Q staged once into LDS; K/V (shared across heads)
// double-buffered per 32-key tile: K natural [32][128] (pad 136), V transposed
// Vt[128][32] (pad 40, packed-b64 staging).  Online softmax: row max via
// shfl_xor over 16-lane halves; row SUM via an extra WMMA against an all-ones
// B fragment (P @ 1 -> sum in every accumulator column, inherits the online
// rescale) -- no sum shuffles.  P re-layouts through per-wave LDS
// (explicit s_wait_dscnt) into an A fragment for P@V.
// ---------------------------------------------------------------------------
__global__ __launch_bounds__(256, 1)
void mqa_flash_attn(const unsigned short* __restrict__ Q,
                    const unsigned short* __restrict__ Kb,
                    const unsigned short* __restrict__ Vb,
                    unsigned short* __restrict__ O) {
    __shared__ unsigned short Qs[128 * 136];
    __shared__ unsigned short Ks[2][32 * 136];
    __shared__ unsigned short Vt[2][128 * 40];
    __shared__ unsigned short Ps[8 * 16 * 40];

    const int tid  = threadIdx.x;
    const int lane = tid & 31, wave = tid >> 5;
    const int half = lane >> 4, l15 = lane & 15;
    const int qt = blockIdx.x, h = blockIdx.y, b = blockIdx.z;
    const size_t tok0 = (size_t)b * SEQ;

    // ---- stage Q tile [128 queries x 128 dk] into LDS (once) ----
    {
        const int row = tid >> 1, seg = tid & 1;
        const v4u* g = reinterpret_cast<const v4u*>(
            Q + (tok0 + qt * 128 + row) * (size_t)D_MODEL + (size_t)h * D_KH + seg * 64);
        v4u* s = reinterpret_cast<v4u*>(&Qs[row * 136 + seg * 64]);
        #pragma unroll
        for (int j = 0; j < 8; j++) s[j] = g[j];
    }

    // o[0..7]: attn output tiles; o[8]: probability row-sum (ones column trick)
    v8f o[9];
    for (int i = 0; i < 9; i++)
        for (int r = 0; r < 8; r++) o[i][r] = 0.f;
    float mprev[8];
    for (int r = 0; r < 8; r++) mprev[r] = -3.0e38f;
    const float sc = 0.088388347648318447f;   // 1/sqrt(128)

    ABFrag ones;                              // bf16 1.0 x 16
    {
        v4u one4 = {0x3F803F80u, 0x3F803F80u, 0x3F803F80u, 0x3F803F80u};
        ones.q[0] = one4; ones.q[1] = one4;
    }

    const int skey = tid >> 3, sseg = tid & 7;        // K staging (natural)
    const int vdk  = tid & 31, vkq  = tid >> 5;       // V staging (4dk x 4key)
    const unsigned short* Kbase = Kb + (tok0 + skey) * (size_t)D_KH + sseg * 16;
    const unsigned short* Vbase = Vb + tok0 * (size_t)D_KH + vdk * 4;

    v4u  k0, k1;
    Pack4 vrow[4];
    auto loadKV = [&](int kv0) {
        const v4u* gk = reinterpret_cast<const v4u*>(Kbase + (size_t)kv0 * D_KH);
        k0 = gk[0]; k1 = gk[1];
        #pragma unroll
        for (int kk = 0; kk < 4; kk++)
            vrow[kk].d = *reinterpret_cast<const v2u*>(
                Vbase + (size_t)(kv0 + vkq * 4 + kk) * D_KH);
    };
    auto storeKV = [&](int buf) {
        v4u* s = reinterpret_cast<v4u*>(&Ks[buf][skey * 136 + sseg * 16]);
        s[0] = k0; s[1] = k1;
        #pragma unroll
        for (int i = 0; i < 4; i++) {   // pack 4 keys (contiguous in Vt row)
            Pack4 w;
            w.u[0] = vrow[0].u[i]; w.u[1] = vrow[1].u[i];
            w.u[2] = vrow[2].u[i]; w.u[3] = vrow[3].u[i];
            *reinterpret_cast<v2u*>(&Vt[buf][(vdk * 4 + i) * 40 + vkq * 4]) = w.d;
        }
    };

    loadKV(0);
    storeKV(0);
    __syncthreads();

    const int niter = SEQ / 32;
    for (int it = 0; it < niter; it++) {
        const int cur = it & 1;
        const bool hasNext = (it + 1) < niter;
        if (hasNext) loadKV((it + 1) * 32);
        if (it + 2 < niter) {
            __builtin_prefetch(Kbase + (size_t)(it + 2) * 32 * D_KH, 0, 0);
            __builtin_prefetch(Vbase + (size_t)((it + 2) * 32 + vkq * 4) * D_KH, 0, 0);
        }

        // ---- S = Q K^T : 16 x 32 in two 16x16 accumulators ----
        v8f s0, s1;
        for (int r = 0; r < 8; r++) { s0[r] = 0.f; s1[r] = 0.f; }
        #pragma unroll
        for (int c = 0; c < 4; c++) {
            ABFrag aq, bk0, bk1;
            const int qm = wave * 16 + l15;
            aq.q[0] = *reinterpret_cast<const v4u*>(&Qs[qm * 136 + c * 32 + 8 * half]);
            aq.q[1] = *reinterpret_cast<const v4u*>(&Qs[qm * 136 + c * 32 + 16 + 8 * half]);
            const v4u* p0 = reinterpret_cast<const v4u*>(&Ks[cur][l15 * 136 + c * 32 + 8 * half]);
            bk0.q[0] = p0[0]; bk0.q[1] = p0[2];
            const v4u* p1 = reinterpret_cast<const v4u*>(&Ks[cur][(16 + l15) * 136 + c * 32 + 8 * half]);
            bk1.q[0] = p1[0]; bk1.q[1] = p1[2];
            s0 = wmma_bf16(aq, bk0, s0);
            s1 = wmma_bf16(aq, bk1, s1);
        }

        // ---- online softmax: row max via shfl_xor; P written to LDS ----
        float corr[8];
        #pragma unroll
        for (int r = 0; r < 8; r++) {
            const float a0 = s0[r] * sc, a1 = s1[r] * sc;
            float m = fmaxf(a0, a1);
            #pragma unroll
            for (int off = 1; off < 16; off <<= 1)
                m = fmaxf(m, __shfl_xor(m, off, 32));
            const float mnew = fmaxf(mprev[r], m);
            corr[r] = __expf(mprev[r] - mnew);
            mprev[r] = mnew;
            const float p0 = __expf(a0 - mnew);
            const float p1 = __expf(a1 - mnew);
            Ps[wave * 640 + (r + 8 * half) * 40 + l15]      = f2bf(p0);
            Ps[wave * 640 + (r + 8 * half) * 40 + 16 + l15] = f2bf(p1);
        }
        #pragma unroll
        for (int i = 0; i < 9; i++)
            #pragma unroll
            for (int r = 0; r < 8; r++) o[i][r] *= corr[r];

        // wave-local LDS RAW: P stores -> A-fragment loads
        asm volatile("s_wait_dscnt 0" ::: "memory");

        ABFrag ap;
        ap.q[0] = *reinterpret_cast<const v4u*>(&Ps[wave * 640 + l15 * 40 + 8 * half]);
        ap.q[1] = *reinterpret_cast<const v4u*>(&Ps[wave * 640 + l15 * 40 + 16 + 8 * half]);

        #pragma unroll
        for (int ni = 0; ni < 8; ni++) {
            ABFrag bv;
            const v4u* p = reinterpret_cast<const v4u*>(&Vt[cur][(ni * 16 + l15) * 40 + 8 * half]);
            bv.q[0] = p[0]; bv.q[1] = p[2];
            o[ni] = wmma_bf16(ap, bv, o[ni]);
        }
        o[8] = wmma_bf16(ap, ones, o[8]);    // row-sum column

        if (hasNext) storeKV(cur ^ 1);
        __syncthreads();
    }

    // finalize: divide by softmax sum (o[8] holds it in every column)
    float rinv[8];
    #pragma unroll
    for (int r = 0; r < 8; r++) rinv[r] = 1.f / o[8][r];
    const size_t obase = (tok0 + qt * 128 + wave * 16) * (size_t)D_MODEL + (size_t)h * D_KH;
    #pragma unroll
    for (int ni = 0; ni < 8; ni++)
        #pragma unroll
        for (int r = 0; r < 8; r++)
            O[obase + (size_t)(r + 8 * half) * D_MODEL + ni * 16 + l15] =
                f2bf(o[ni][r] * rinv[r]);
}

// ---------------------------------------------------------------------------
extern "C" void kernel_launch(void* const* d_in, const int* in_sizes, int n_in,
                              void* d_out, int out_size, void* d_ws, size_t ws_size,
                              hipStream_t stream) {
    (void)in_sizes; (void)n_in; (void)out_size; (void)ws_size;
    const float* x  = (const float*)d_in[0];
    const float* sA = (const float*)d_in[1];
    const float* sF = (const float*)d_in[2];
    const float* Wq = (const float*)d_in[3];
    const float* bq = (const float*)d_in[4];
    const float* Wk = (const float*)d_in[5];
    const float* bk = (const float*)d_in[6];
    const float* Wv = (const float*)d_in[7];
    const float* bv = (const float*)d_in[8];
    const float* Wo = (const float*)d_in[9];
    const float* bo = (const float*)d_in[10];
    const float* W1 = (const float*)d_in[11];
    const float* b1 = (const float*)d_in[12];
    const float* W2 = (const float*)d_in[13];
    const float* b2 = (const float*)d_in[14];
    float* out = (float*)d_out;

    // workspace carve-out (~575 MB total)
    char* ws = (char*)d_ws;
    size_t off = 0;
    auto alloc = [&](size_t bytes) -> unsigned short* {
        unsigned short* p = (unsigned short*)(ws + off);
        off = (off + bytes + 255) & ~(size_t)255;
        return p;
    };
    unsigned short* WqB = alloc((size_t)D_MODEL * D_MODEL * 2);
    unsigned short* WkB = alloc((size_t)D_MODEL * D_KH * 2);
    unsigned short* WvB = alloc((size_t)D_MODEL * D_KH * 2);
    unsigned short* WoB = alloc((size_t)D_MODEL * D_MODEL * 2);
    unsigned short* W1B = alloc((size_t)D_MODEL * D_FF * 2);
    unsigned short* W2B = alloc((size_t)D_FF * D_MODEL * 2);
    unsigned short* xnB = alloc((size_t)NTOK * D_MODEL * 2);
    unsigned short* qB  = alloc((size_t)NTOK * D_MODEL * 2);
    unsigned short* kB  = alloc((size_t)NTOK * D_KH * 2);
    unsigned short* vB  = alloc((size_t)NTOK * D_KH * 2);
    unsigned short* atB = alloc((size_t)NTOK * D_MODEL * 2);
    unsigned short* hB  = alloc((size_t)NTOK * D_FF * 2);

    auto cvt = [&](const float* src, unsigned short* dst, size_t n) {
        int n4 = (int)(n / 4);
        cvt_f32_to_bf16<<<(n4 + 255) / 256, 256, 0, stream>>>(src, dst, n4);
    };
    cvt(Wq, WqB, (size_t)D_MODEL * D_MODEL);
    cvt(Wk, WkB, (size_t)D_MODEL * D_KH);
    cvt(Wv, WvB, (size_t)D_MODEL * D_KH);
    cvt(Wo, WoB, (size_t)D_MODEL * D_MODEL);
    cvt(W1, W1B, (size_t)D_MODEL * D_FF);
    cvt(W2, W2B, (size_t)D_FF * D_MODEL);

    // ---- attention branch ----
    rmsnorm_to_bf16<<<NTOK, 256, 0, stream>>>(x, sA, xnB);

    gemm_bf16_wmma<0, 0><<<dim3(D_MODEL / 128, NTOK / 128), 256, 0, stream>>>(
        xnB, WqB, bq, nullptr, nullptr, qB, NTOK, D_MODEL, D_MODEL);
    gemm_bf16_wmma<0, 0><<<dim3(D_KH / 128, NTOK / 128), 256, 0, stream>>>(
        xnB, WkB, bk, nullptr, nullptr, kB, NTOK, D_KH, D_MODEL);
    gemm_bf16_wmma<0, 0><<<dim3(D_KH / 128, NTOK / 128), 256, 0, stream>>>(
        xnB, WvB, bv, nullptr, nullptr, vB, NTOK, D_KH, D_MODEL);

    mqa_flash_attn<<<dim3(SEQ / 128, N_HEADS, 2), 256, 0, stream>>>(qB, kB, vB, atB);

    // out = x + attn @ Wo + bo
    gemm_bf16_wmma<1, 0><<<dim3(D_MODEL / 128, NTOK / 128), 256, 0, stream>>>(
        atB, WoB, bo, x, out, nullptr, NTOK, D_MODEL, D_MODEL);

    // ---- FFN branch ----
    rmsnorm_to_bf16<<<NTOK, 256, 0, stream>>>(out, sF, xnB);

    gemm_bf16_wmma<0, 1><<<dim3(D_FF / 128, NTOK / 128), 256, 0, stream>>>(
        xnB, W1B, b1, nullptr, nullptr, hB, NTOK, D_FF, D_MODEL);

    // out = out + h @ W2 + b2
    gemm_bf16_wmma<1, 0><<<dim3(D_MODEL / 128, NTOK / 128), 256, 0, stream>>>(
        hB, W2B, b2, out, out, nullptr, NTOK, D_MODEL, D_FF);
}
